// SelfAttention_51410758533751
// MI455X (gfx1250) — matrix operand order repaired and checked by reading, not compiled, not run on hardware
//
#include <hip/hip_runtime.h>
#include <hip/hip_bf16.h>

// Problem constants (match reference)
#define BB 8
#define CC 512
#define C8 64
#define WW 2048
#define KB 256     // attention: key block per iteration
#define NB 32      // proj GEMM: positions (N) per block
#define MALL 640   // combined output channels: 64 q + 64 k + 512 v
#define XPAD 520   // LDS row pad: stride 1040B == 260 dwords == 4 (mod 64 banks)

typedef __attribute__((ext_vector_type(16))) __bf16 v16bf;
typedef __attribute__((ext_vector_type(8)))  float  v8f;

union BF16Tile {
    v16bf v;
    unsigned int u[8];
};

__device__ __forceinline__ unsigned short f32_to_bf16(float f) {
    unsigned int u = __builtin_bit_cast(unsigned int, f);
    unsigned int r = u + 0x7FFFu + ((u >> 16) & 1u);   // round-to-nearest-even
    return (unsigned short)(r >> 16);
}

// ---------------------------------------------------------------------------
// Pack Wq/Wk/Wv into one bf16 A-operand matrix Wall[640][512] (row-major, so
// the WMMA A-layout K-pairs are contiguous) and the biases into ball[640].
// ---------------------------------------------------------------------------
__global__ __launch_bounds__(256) void pack_weights_kernel(
    const float* __restrict__ Wq, const float* __restrict__ bq,
    const float* __restrict__ Wk, const float* __restrict__ bk,
    const float* __restrict__ Wv, const float* __restrict__ bv,
    unsigned short* __restrict__ Wall, float* __restrict__ ball) {
    const int o = blockIdx.x;                  // 0..639
    const float* src;
    float bias;
    if (o < 64)       { src = Wq + (size_t)o * CC;         bias = bq[o]; }
    else if (o < 128) { src = Wk + (size_t)(o - 64) * CC;  bias = bk[o - 64]; }
    else              { src = Wv + (size_t)(o - 128) * CC; bias = bv[o - 128]; }
    for (int c = threadIdx.x; c < CC; c += 256)
        Wall[(size_t)o * CC + c] = f32_to_bf16(src[c]);
    if (threadIdx.x == 0) ball[o] = bias;
}

// ---------------------------------------------------------------------------
// Fused q/k/v projection as a WMMA GEMM: D[640 x 32] = Wall[640 x 512] @
// x[b, 512, w0:w0+32].  x strip staged in LDS as bf16 [w][c] so the B-operand
// K(=c_in)-pairs are contiguous (ds_load_b128).  8 waves x 5 M-tiles x 2
// N-tiles; 160 v_wmma_f32_16x16x32_bf16 per wave.
// ---------------------------------------------------------------------------
__global__ __launch_bounds__(256) void proj_gemm_kernel(
    const float* __restrict__ x,
    const unsigned short* __restrict__ Wall, const float* __restrict__ ball,
    unsigned short* __restrict__ qb, unsigned short* __restrict__ kb,
    unsigned short* __restrict__ vb) {
    __shared__ __align__(16) unsigned short sX[NB][XPAD];

    const int tid  = threadIdx.x;
    const int wave = tid >> 5;
    const int lane = tid & 31;
    const int n    = lane & 15;
    const int half = lane >> 4;

    const int bw = blockIdx.x;
    const int b  = bw >> 6;              // W/NB = 64 strips per batch
    const int w0 = (bw & 63) * NB;

    // Stage x[b, :, w0:w0+32] -> LDS bf16 [w_local][c]; coalesced f32 reads.
    for (int idx = tid; idx < NB * CC; idx += 256) {
        const int c  = idx >> 5;         // NB == 32
        const int wl = idx & 31;
        sX[wl][c] = f32_to_bf16(x[((size_t)b * CC + c) * WW + w0 + wl]);
    }
    __syncthreads();

    v8f acc[5][2];
    #pragma unroll
    for (int t = 0; t < 5; ++t)
        #pragma unroll
        for (int nt = 0; nt < 2; ++nt)
            acc[t][nt] = (v8f){0.f, 0.f, 0.f, 0.f, 0.f, 0.f, 0.f, 0.f};

    #pragma unroll 4
    for (int s = 0; s < 16; ++s) {       // K = 512 = 16 x 32
        BF16Tile Bt[2];
        #pragma unroll
        for (int nt = 0; nt < 2; ++nt)
            #pragma unroll
            for (int v = 0; v < 8; ++v)  // B 32x16: lane n=col, K pair 2v,2v+1
                Bt[nt].u[v] = *(const unsigned int*)(
                    &sX[nt * 16 + n][s * 32 + (v >> 2) * 16 + half * 8 + (v & 3) * 2]);

        #pragma unroll
        for (int t = 0; t < 5; ++t) {
            const int crow = (wave * 5 + t) * 16 + n;   // A row m = lane%16
            const unsigned short* wrow = Wall + (size_t)crow * CC + s * 32;
            BF16Tile At;
            #pragma unroll
            for (int v = 0; v < 8; ++v) {
                const int k = (v >> 2) * 16 + half * 8 + (v & 3) * 2;  // A 16x32
                At.u[v] = *(const unsigned int*)(wrow + k);
            }
            #pragma unroll
            for (int nt = 0; nt < 2; ++nt)
                acc[t][nt] = __builtin_amdgcn_wmma_f32_16x16x32_bf16(
                    false, At.v, false, Bt[nt].v, (short)0, acc[t][nt],
                    false, false);
        }
    }

    // Epilogue: bias add, bf16 convert, scatter to q/k/v layouts.
    #pragma unroll
    for (int t = 0; t < 5; ++t) {
        #pragma unroll
        for (int r = 0; r < 8; ++r) {
            const int co = (wave * 5 + t) * 16 + half * 8 + r;  // D row m
            const float bias = ball[co];
            #pragma unroll
            for (int nt = 0; nt < 2; ++nt) {
                const int w = w0 + nt * 16 + n;                 // D col n
                const unsigned short h = f32_to_bf16(acc[t][nt][r] + bias);
                if (co < 64)
                    qb[((size_t)b * WW + w) * C8 + co] = h;
                else if (co < 128)
                    kb[((size_t)b * WW + w) * C8 + (co - 64)] = h;
                else
                    vb[((size_t)b * CC + (co - 128)) * WW + w] = h;
            }
        }
    }
}

// ---------------------------------------------------------------------------
// Flash-style attention: one 256-thread block (8 wave32) per 16-query tile.
// S and P strips live in LDS; both GEMMs use v_wmma_f32_16x16x32_bf16.
// ---------------------------------------------------------------------------
__global__ __launch_bounds__(256) void attn_kernel(
    const unsigned short* __restrict__ qb, const unsigned short* __restrict__ kb,
    const unsigned short* __restrict__ vb, const float* __restrict__ x,
    const float* __restrict__ gamma, float* __restrict__ out) {
    __shared__ float sS[16][KB];                          // scores f32 [query][key]
    __shared__ __align__(16) unsigned short sP[16][KB];   // probs bf16 [query][key]
    __shared__ float sM[16], sL[16], sA[16];

    const int tid  = threadIdx.x;
    const int wave = tid >> 5;
    const int lane = tid & 31;
    const int n    = lane & 15;    // tile column (key in S, query in out)
    const int half = lane >> 4;

    const int blk = blockIdx.x;
    const int b   = blk >> 7;               // 128 query tiles per batch
    const int w0  = (blk & 127) << 4;       // query base

    const float g = gamma[0];

    // Preload Q A-operands (16 queries x 64 d) as two 16x32 bf16 tiles.
    BF16Tile qA[2];
    {
        const size_t qbase = ((size_t)b * WW + (w0 + n)) * C8;  // A row m = lane%16
        #pragma unroll
        for (int s = 0; s < 2; ++s) {
            #pragma unroll
            for (int v = 0; v < 8; ++v) {
                const int k = (v >> 2) * 16 + half * 8 + (v & 3) * 2;  // A 16x32 layout
                qA[s].u[v] = *(const unsigned int*)(qb + qbase + s * 32 + k);
            }
        }
    }

    if (tid < 16) { sM[tid] = -1e30f; sL[tid] = 0.f; sA[tid] = 1.f; }
    __syncthreads();

    // Output accumulators: this wave owns channels [wave*64, wave*64+64).
    v8f acc[4];
    #pragma unroll
    for (int t = 0; t < 4; ++t) acc[t] = (v8f){0.f, 0.f, 0.f, 0.f, 0.f, 0.f, 0.f, 0.f};

    for (int j0 = 0; j0 < WW; j0 += KB) {
        // ---- Phase 1: S[0:16][j0:j0+KB] = Q Kt, 2 key tiles per wave ----
        #pragma unroll
        for (int jt2 = 0; jt2 < 2; ++jt2) {
            const int jt = wave * 2 + jt2;                 // 0..15
            const int j  = j0 + jt * 16 + n;               // this lane's key column
            const size_t kbase = ((size_t)b * WW + j) * C8;
            v8f d = (v8f){0.f, 0.f, 0.f, 0.f, 0.f, 0.f, 0.f, 0.f};
            #pragma unroll
            for (int s = 0; s < 2; ++s) {
                BF16Tile kB;
                #pragma unroll
                for (int v = 0; v < 8; ++v) {
                    const int dd = s * 32 + (v >> 2) * 16 + half * 8 + (v & 3) * 2;  // B 32x16 layout
                    kB.u[v] = *(const unsigned int*)(kb + kbase + dd);
                }
                d = __builtin_amdgcn_wmma_f32_16x16x32_bf16(
                        false, qA[s].v, false, kB.v, (short)0, d, false, false);
            }
            #pragma unroll
            for (int r = 0; r < 8; ++r)                    // D row m = half*8 + r
                sS[half * 8 + r][jt * 16 + n] = d[r];
        }
        __syncthreads();

        // ---- Phase 2: online softmax over this key block (2 rows per wave) ----
        #pragma unroll
        for (int rr = 0; rr < 2; ++rr) {
            const int i = wave * 2 + rr;
            float vals[8];
            float vmax = -1e30f;
            #pragma unroll
            for (int u = 0; u < 8; ++u) {
                vals[u] = sS[i][lane + 32 * u];
                vmax = fmaxf(vmax, vals[u]);
            }
            #pragma unroll
            for (int off = 16; off > 0; off >>= 1)
                vmax = fmaxf(vmax, __shfl_xor(vmax, off, 32));
            const float oldm = sM[i];
            const float newm = fmaxf(oldm, vmax);
            float sum = 0.f;
            #pragma unroll
            for (int u = 0; u < 8; ++u) {
                const float p = __expf(vals[u] - newm);
                sum += p;
                sP[i][lane + 32 * u] = f32_to_bf16(p);
            }
            #pragma unroll
            for (int off = 16; off > 0; off >>= 1)
                sum += __shfl_xor(sum, off, 32);
            if (lane == 0) {
                const float alpha = __expf(oldm - newm);
                sA[i] = alpha;
                sM[i] = newm;
                sL[i] = sL[i] * alpha + sum;
            }
        }
        __syncthreads();

        // ---- Phase 3: rescale accumulators, accumulate V @ P ----
        const float alpha = sA[n];        // per output column (query) rescale
        #pragma unroll
        for (int t = 0; t < 4; ++t)
            #pragma unroll
            for (int r = 0; r < 8; ++r) acc[t][r] *= alpha;

        const int cbase = wave * 64;
        #pragma unroll
        for (int kk = 0; kk < KB; kk += 32) {
            BF16Tile pB;                  // P block: (32 keys) x (16 queries)
            #pragma unroll
            for (int v = 0; v < 8; ++v) {
                const int key = kk + (v >> 2) * 16 + half * 8 + (v & 3) * 2;
                pB.u[v] = *(const unsigned int*)(&sP[n][key]);
            }
            #pragma unroll
            for (int t = 0; t < 4; ++t) {
                const int c = cbase + t * 16 + n;           // A row = channel
                const size_t vbase = ((size_t)b * CC + c) * WW + j0;
                if (kk == 0 && j0 + KB < WW)
                    __builtin_prefetch(vb + vbase + KB, 0, 1);  // global_prefetch_b8
                BF16Tile vA;
                #pragma unroll
                for (int v = 0; v < 8; ++v) {
                    const int k = (v >> 2) * 16 + half * 8 + (v & 3) * 2;
                    vA.u[v] = *(const unsigned int*)(vb + vbase + kk + k);
                }
                acc[t] = __builtin_amdgcn_wmma_f32_16x16x32_bf16(
                             false, vA.v, false, pB.v, (short)0, acc[t], false, false);
            }
        }
    }

    // ---- Epilogue: out = gamma * (acc / l) + x ----
    const float invl = 1.0f / sL[n];
    const int w = w0 + n;
    #pragma unroll
    for (int t = 0; t < 4; ++t) {
        #pragma unroll
        for (int r = 0; r < 8; ++r) {
            const int c = wave * 64 + t * 16 + half * 8 + r;
            const size_t idx = ((size_t)b * CC + c) * WW + w;
            out[idx] = g * (acc[t][r] * invl) + x[idx];
        }
    }
}

extern "C" void kernel_launch(void* const* d_in, const int* in_sizes, int n_in,
                              void* d_out, int out_size, void* d_ws, size_t ws_size,
                              hipStream_t stream) {
    const float* x     = (const float*)d_in[0];
    const float* Wq    = (const float*)d_in[1];
    const float* bq    = (const float*)d_in[2];
    const float* Wk    = (const float*)d_in[3];
    const float* bk    = (const float*)d_in[4];
    const float* Wv    = (const float*)d_in[5];
    const float* bv    = (const float*)d_in[6];
    const float* gamma = (const float*)d_in[7];
    float* out = (float*)d_out;

    // Workspace: qb (2 MB) | kb (2 MB) | vb (16 MB) | Wall (640 KB) | ball (2.5 KB)
    unsigned short* qb   = (unsigned short*)d_ws;
    unsigned short* kb   = qb + (size_t)BB * WW * C8;
    unsigned short* vb   = kb + (size_t)BB * WW * C8;
    unsigned short* Wall = vb + (size_t)BB * CC * WW;
    float*          ball = (float*)(Wall + (size_t)MALL * CC);

    pack_weights_kernel<<<MALL, 256, 0, stream>>>(Wq, bq, Wk, bk, Wv, bv, Wall, ball);
    proj_gemm_kernel<<<BB * (WW / NB), 256, 0, stream>>>(x, Wall, ball, qb, kb, vb);
    attn_kernel<<<BB * (WW / 16), 256, 0, stream>>>(qb, kb, vb, x, gamma, out);
}
